// ECGHybridModel_66262755442979
// MI455X (gfx1250) — matrix operand order repaired and checked by reading, not compiled, not run on hardware
//
#include <hip/hip_runtime.h>
#include <hip/hip_bf16.h>

typedef __attribute__((ext_vector_type(16))) _Float16 v16h;
typedef __attribute__((ext_vector_type(8)))  float    v8f;

#define BTOT 65536
#define TA_THREADS 128

// CDNA5 hardware transcendental: V_TANH_F32 (TRANS32 class).
__device__ __forceinline__ float htanh(float x) {
    float r;
    asm volatile("v_tanh_f32 %0, %1" : "=v"(r) : "v"(x));
    return r;
}
// sigmoid(x) == 0.5*tanh(x/2) + 0.5  -> one trans op + one FMA
__device__ __forceinline__ float hsig(float x) {
    return __builtin_fmaf(0.5f, htanh(0.5f * x), 0.5f);
}

// ---------------------------------------------------------------------------
// Kernel A: conv1+bn1+relu -> conv2+bn2+relu -> LIF1 -> LIF2 (bit-packed
// spikes) + spatial average.  One thread per sample.
// ---------------------------------------------------------------------------
__global__ void __launch_bounds__(TA_THREADS) frontend_kernel(
    const float* __restrict__ x,
    const float* __restrict__ conv1_w, const float* __restrict__ conv1_b,
    const float* __restrict__ bn1_g, const float* __restrict__ bn1_b,
    const float* __restrict__ bn1_m, const float* __restrict__ bn1_v,
    const float* __restrict__ conv2_w, const float* __restrict__ conv2_b,
    const float* __restrict__ bn2_g, const float* __restrict__ bn2_b,
    const float* __restrict__ bn2_m, const float* __restrict__ bn2_v,
    unsigned long long* __restrict__ spike_bits,
    float* __restrict__ spatial_avg)
{
    __shared__ float w1f[6][5];
    __shared__ float b1f[6];
    __shared__ float w2f[12][6][3];
    __shared__ float b2f[12];
    __shared__ float mem1[33][TA_THREADS];
    __shared__ float mem2[33][TA_THREADS];

    const int tid = threadIdx.x;

    // Fold BN (and conv bias) into effective conv scale/bias once per block.
    for (int idx = tid; idx < 264; idx += TA_THREADS) {
        if (idx < 30) {
            int c = idx / 5, k = idx % 5;
            float s = bn1_g[c] * rsqrtf(bn1_v[c] + 1e-5f);
            w1f[c][k] = conv1_w[c * 5 + k] * s;
        } else if (idx < 36) {
            int c = idx - 30;
            float s = bn1_g[c] * rsqrtf(bn1_v[c] + 1e-5f);
            b1f[c] = conv1_b[c] * s + bn1_b[c] - bn1_m[c] * s;
        } else if (idx < 252) {
            int j = idx - 36;
            int ch = j / 18;
            float s = bn2_g[ch] * rsqrtf(bn2_v[ch] + 1e-5f);
            ((float*)w2f)[j] = conv2_w[j] * s;
        } else {
            int ch = idx - 252;
            float s = bn2_g[ch] * rsqrtf(bn2_v[ch] + 1e-5f);
            b2f[ch] = conv2_b[ch] * s + bn2_b[ch] - bn2_m[ch] * s;
        }
    }
    for (int k = 0; k < 33; ++k) { mem1[k][tid] = 0.f; mem2[k][tid] = 0.f; }
    __syncthreads();

    const int b = blockIdx.x * TA_THREADS + tid;
    const float* xr = x + (size_t)b * 360;

    float sacc[12];
#pragma unroll
    for (int c = 0; c < 12; ++c) sacc[c] = 0.f;

    int kcur = 0, tcur = 0;
    unsigned long long bits = 0ull;

    for (int p = 0; p < 90; ++p) {
        // x window: indices 4p-4 .. 4p+4 (9 values), zero-padded.
        float xv[9];
        const int base = 4 * p - 4;
#pragma unroll
        for (int i = 0; i < 9; ++i) {
            int gi = base + i;
            xv[i] = (gi >= 0 && gi < 360) ? xr[gi] : 0.f;
        }
        // conv1+bn1+relu at q = 2p-1+dq  (x idx = 2q-2+kx = base + 2dq + kx)
        float h1v[3][6];
#pragma unroll
        for (int dq = 0; dq < 3; ++dq) {
            const int q = 2 * p - 1 + dq;
            const bool valid = (q >= 0);  // q < 180 always holds
#pragma unroll
            for (int c = 0; c < 6; ++c) {
                float a = b1f[c];
#pragma unroll
                for (int kx = 0; kx < 5; ++kx) a += xv[2 * dq + kx] * w1f[c][kx];
                a = fmaxf(a, 0.f);
                h1v[dq][c] = valid ? a : 0.f;
            }
        }
        // conv2+bn2+relu -> 12 channels at position p, then stream into LIFs
#pragma unroll
        for (int ch = 0; ch < 12; ++ch) {
            float a = b2f[ch];
#pragma unroll
            for (int dq = 0; dq < 3; ++dq)
#pragma unroll
                for (int c = 0; c < 6; ++c) a += h1v[dq][c] * w2f[ch][c][dq];
            float h2 = fmaxf(a, 0.f);
            sacc[ch] += h2;
            if (tcur < 32) {
                float m1 = 0.95f * mem1[kcur][tid] + h2;
                float s1 = (m1 > 0.5f) ? 1.f : 0.f;
                mem1[kcur][tid] = m1 * (1.f - s1);
                float m2 = 0.9f * mem2[kcur][tid] + s1;
                float s2 = (m2 > 0.6f) ? 1.f : 0.f;
                mem2[kcur][tid] = m2 * (1.f - s2);
                if (s2 > 0.5f) bits |= (1ull << kcur);
                if (++kcur == 33) {
                    spike_bits[(size_t)b * 32 + tcur] = bits;
                    bits = 0ull; kcur = 0; ++tcur;
                }
            }
        }
    }
#pragma unroll
    for (int c = 0; c < 12; ++c)
        spatial_avg[(size_t)b * 12 + c] = sacc[c] * (1.f / 90.f);
}

// ---------------------------------------------------------------------------
// Kernel B: LSTM over 32 steps via v_wmma_f32_16x16x32_f16 + fused head.
// One wave (32 lanes) handles a 16-sample batch tile.
// A operand per step = [spikes(33) | 0(7) | h(24)] padded to K=64.
// ---------------------------------------------------------------------------
__global__ void __launch_bounds__(32) lstm_head_kernel(
    const unsigned long long* __restrict__ spike_bits,
    const float* __restrict__ spatial_avg,
    const float* __restrict__ wih, const float* __restrict__ whh,
    const float* __restrict__ bih, const float* __restrict__ bhh,
    const float* __restrict__ rt_w1, const float* __restrict__ rt_b1,
    const float* __restrict__ rt_w2, const float* __restrict__ rt_b2,
    const float* __restrict__ fus_w, const float* __restrict__ fus_b,
    const float* __restrict__ cls_w1, const float* __restrict__ cls_b1,
    const float* __restrict__ cls_w2, const float* __restrict__ cls_b2,
    float* __restrict__ out)
{
    __shared__ float    glds[16][96];   // gate staging tile (M x 4H)
    __shared__ _Float16 h16[16][24];    // hidden state, f16 for A-fragment
    __shared__ float    wl[1723];       // head weights

    const int l   = threadIdx.x;
    const int blk = blockIdx.x;
    const int lm  = l & 15;
    const bool lo = (l < 16);

    // Stage head weights in LDS.
    for (int i = l; i < 1723; i += 32) {
        float v;
        if      (i < 432)  v = rt_w1[i];
        else if (i < 444)  v = rt_b1[i - 432];
        else if (i < 468)  v = rt_w2[i - 444];
        else if (i < 470)  v = rt_b2[i - 468];
        else if (i < 1334) v = fus_w[i - 470];
        else if (i < 1358) v = fus_b[i - 1334];
        else if (i < 1646) v = cls_w1[i - 1358];
        else if (i < 1658) v = cls_b1[i - 1646];
        else if (i < 1718) v = cls_w2[i - 1658];
        else               v = cls_b2[i - 1718];
        wl[i] = v;
    }
    for (int i = l; i < 16 * 24; i += 32) ((_Float16*)h16)[i] = (_Float16)0.f;

    // B fragments: combined W[g][K], K<33 -> Wih[g][K], K in [40,64) -> Whh[g][K-40].
    // B layout (16-bit, 32x16): VGPR v: lanes0-15 K=2v,2v+1 (v<4), 8+2v,9+2v (v>=4); lanes16-31 K +8; N=lane%16.
    v16h bfrag[6][2];
    float bias[6];
#pragma unroll
    for (int nt = 0; nt < 6; ++nt) {
        const int g = nt * 16 + lm;
        bias[nt] = bih[g] + bhh[g];
#pragma unroll
        for (int kh = 0; kh < 2; ++kh) {
            v16h f;
#pragma unroll
            for (int v = 0; v < 8; ++v) {
#pragma unroll
                for (int j = 0; j < 2; ++j) {
                    const int K = 32 * kh + (lo ? 0 : 8) + 2 * v + j + (v < 4 ? 0 : 8);
                    float w = 0.f;
                    if (K < 33)       w = wih[g * 33 + K];
                    else if (K >= 40) w = whh[g * 24 + (K - 40)];
                    f[2 * v + j] = (_Float16)w;
                }
            }
            bfrag[nt][kh] = f;
        }
    }

    // Cell state: lane covers (m = lm, j in [jb, jb+12)).
    float cst[12];
#pragma unroll
    for (int j = 0; j < 12; ++j) cst[j] = 0.f;
    const int jb = lo ? 0 : 12;

    const unsigned long long* sb = spike_bits + (size_t)(blk * 16 + lm) * 32;
    __syncthreads();

    for (int t = 0; t < 32; ++t) {
        const unsigned long long bits = sb[t];
        // A layout (16-bit, 16x32): lanes0-15: K 0-7 (v0-3), 16-23 (v4-7);
        //                           lanes16-31: K 8-15, 24-31.  M = lane%16.
        v16h a1;
        const int kb1 = lo ? 0 : 8, kb2 = lo ? 16 : 24;
#pragma unroll
        for (int e = 0; e < 8; ++e)
            a1[e] = ((bits >> (kb1 + e)) & 1ull) ? (_Float16)1.f : (_Float16)0.f;
#pragma unroll
        for (int e = 0; e < 8; ++e)
            a1[8 + e] = ((bits >> (kb2 + e)) & 1ull) ? (_Float16)1.f : (_Float16)0.f;

        v16h a2;  // K 32..63: [x32, 0x7, h0..h23]
        if (lo) {
            a2[0] = ((bits >> 32) & 1ull) ? (_Float16)1.f : (_Float16)0.f;
#pragma unroll
            for (int e = 1; e < 8; ++e) a2[e] = (_Float16)0.f;
#pragma unroll
            for (int e = 0; e < 8; ++e) a2[8 + e] = h16[lm][8 + e];   // K 48-55
        } else {
#pragma unroll
            for (int e = 0; e < 8; ++e) a2[e] = h16[lm][e];           // K 40-47
#pragma unroll
            for (int e = 0; e < 8; ++e) a2[8 + e] = h16[lm][16 + e];  // K 56-63
        }

        const int M0 = lo ? 0 : 8;
#pragma unroll
        for (int nt = 0; nt < 6; ++nt) {
            v8f acc;
#pragma unroll
            for (int r = 0; r < 8; ++r) acc[r] = bias[nt];
            acc = __builtin_amdgcn_wmma_f32_16x16x32_f16(
                false, a1, false, bfrag[nt][0], (short)0, acc, false, false);
            acc = __builtin_amdgcn_wmma_f32_16x16x32_f16(
                false, a2, false, bfrag[nt][1], (short)0, acc, false, false);
            // C layout: VGPR r -> M = r (+8 for upper lanes), N = lane%16.
#pragma unroll
            for (int r = 0; r < 8; ++r) glds[M0 + r][nt * 16 + lm] = acc[r];
        }
        __syncthreads();

        // LSTM cell nonlinearity: gates [i | f | g | o] along the 96-dim.
        // sigmoid/tanh via hardware V_TANH_F32 (error << f16 rounding of h).
#pragma unroll
        for (int jj = 0; jj < 12; ++jj) {
            const int j = jb + jj;
            const float gi = glds[lm][j];
            const float gf = glds[lm][24 + j];
            const float gg = glds[lm][48 + j];
            const float go = glds[lm][72 + j];
            float c = hsig(gf) * cst[jj] + hsig(gi) * htanh(gg);
            cst[jj] = c;
            float h = hsig(go) * htanh(c);
            h16[lm][j] = (_Float16)h;
        }
        __syncthreads();
    }

    // Head epilogue: lanes 0..15, one sample each.
    if (lo) {
        const int s = blk * 16 + lm;
        float comb[36];
#pragma unroll
        for (int j = 0; j < 24; ++j) comb[j] = (float)h16[lm][j];
        const float* sp = spatial_avg + (size_t)s * 12;
#pragma unroll
        for (int j = 0; j < 12; ++j) comb[24 + j] = sp[j];

        const float* RT_W1 = wl;
        const float* RT_B1 = wl + 432;
        const float* RT_W2 = wl + 444;
        const float* RT_B2 = wl + 468;
        const float* FUS_W = wl + 470;
        const float* FUS_B = wl + 1334;
        const float* C1W   = wl + 1358;
        const float* C1B   = wl + 1646;
        const float* C2W   = wl + 1658;
        const float* C2B   = wl + 1718;

        float f1[12];
#pragma unroll
        for (int o = 0; o < 12; ++o) {
            float a = RT_B1[o];
#pragma unroll
            for (int j = 0; j < 36; ++j) a += RT_W1[o * 36 + j] * comb[j];
            f1[o] = fmaxf(a, 0.f);
        }
        float z0 = RT_B2[0], z1 = RT_B2[1];
#pragma unroll
        for (int j = 0; j < 12; ++j) { z0 += RT_W2[j] * f1[j]; z1 += RT_W2[12 + j] * f1[j]; }
        // softmax + base reweight + renorm collapses to:
        const float mz = fmaxf(z0, z1);
        const float e0 = 0.7f * expf(z0 - mz);
        const float e1 = 0.3f * expf(z1 - mz);
        const float alpha = e0 / (e0 + e1);

        float fin[36];
#pragma unroll
        for (int j = 0; j < 24; ++j) fin[j] = comb[j] * alpha;
#pragma unroll
        for (int j = 0; j < 12; ++j) fin[24 + j] = comb[24 + j] * (1.f - alpha);

        float fu[24];
#pragma unroll
        for (int o = 0; o < 24; ++o) {
            float a = FUS_B[o];
#pragma unroll
            for (int j = 0; j < 36; ++j) a += FUS_W[o * 36 + j] * fin[j];
            fu[o] = fmaxf(a, 0.f);
        }
        float t1v[12];
#pragma unroll
        for (int o = 0; o < 12; ++o) {
            float a = C1B[o];
#pragma unroll
            for (int j = 0; j < 24; ++j) a += C1W[o * 24 + j] * fu[j];
            t1v[o] = fmaxf(a, 0.f);
        }
#pragma unroll
        for (int o = 0; o < 5; ++o) {
            float a = C2B[o];
#pragma unroll
            for (int j = 0; j < 12; ++j) a += C2W[o * 12 + j] * t1v[j];
            out[(size_t)s * 5 + o] = a;
        }
    }
}

// ---------------------------------------------------------------------------
extern "C" void kernel_launch(void* const* d_in, const int* in_sizes, int n_in,
                              void* d_out, int out_size, void* d_ws, size_t ws_size,
                              hipStream_t stream) {
    const float* x       = (const float*)d_in[0];
    const float* conv1_w = (const float*)d_in[1];
    const float* conv1_b = (const float*)d_in[2];
    const float* bn1_g   = (const float*)d_in[3];
    const float* bn1_b   = (const float*)d_in[4];
    const float* bn1_m   = (const float*)d_in[5];
    const float* bn1_v   = (const float*)d_in[6];
    const float* conv2_w = (const float*)d_in[7];
    const float* conv2_b = (const float*)d_in[8];
    const float* bn2_g   = (const float*)d_in[9];
    const float* bn2_b   = (const float*)d_in[10];
    const float* bn2_m   = (const float*)d_in[11];
    const float* bn2_v   = (const float*)d_in[12];
    const float* wih     = (const float*)d_in[13];
    const float* whh     = (const float*)d_in[14];
    const float* bih     = (const float*)d_in[15];
    const float* bhh     = (const float*)d_in[16];
    const float* rt_w1   = (const float*)d_in[17];
    const float* rt_b1   = (const float*)d_in[18];
    const float* rt_w2   = (const float*)d_in[19];
    const float* rt_b2   = (const float*)d_in[20];
    const float* fus_w   = (const float*)d_in[21];
    const float* fus_b   = (const float*)d_in[22];
    const float* cls_w1  = (const float*)d_in[23];
    const float* cls_b1  = (const float*)d_in[24];
    const float* cls_w2  = (const float*)d_in[25];
    const float* cls_b2  = (const float*)d_in[26];

    unsigned long long* spike_bits = (unsigned long long*)d_ws;
    float* spatial = (float*)((char*)d_ws + (size_t)BTOT * 32 * sizeof(unsigned long long));

    frontend_kernel<<<BTOT / TA_THREADS, TA_THREADS, 0, stream>>>(
        x, conv1_w, conv1_b, bn1_g, bn1_b, bn1_m, bn1_v,
        conv2_w, conv2_b, bn2_g, bn2_b, bn2_m, bn2_v,
        spike_bits, spatial);

    lstm_head_kernel<<<BTOT / 16, 32, 0, stream>>>(
        spike_bits, spatial, wih, whh, bih, bhh,
        rt_w1, rt_b1, rt_w2, rt_b2, fus_w, fus_b,
        cls_w1, cls_b1, cls_w2, cls_b2, (float*)d_out);
}
